// MambaStack1_21036749816515
// MI455X (gfx1250) — matrix operand order repaired and checked by reading, not compiled, not run on hardware
//
#include <hip/hip_runtime.h>
#include <hip/hip_bf16.h>
#include <hip/hip_fp16.h>

#define D_MODEL 256
#define D_STATE 16
#define D_CONV  4
#define D_INNER 512
#define DT_RANK 16
#define N_LAYERS 2
#define N_EXP   2
#define D_FF    256
#define BATCH   4
#define SEQ     2048
#define NTOK    (BATCH * SEQ)   /* 8192 tokens */

typedef __attribute__((ext_vector_type(16))) _Float16 v16h;
typedef __attribute__((ext_vector_type(8)))  _Float16 v8h;
typedef __attribute__((ext_vector_type(8)))  float    v8f;

// ---------------------------------------------------------------------------
// WMMA GEMM:  O[M,N] = A[M,K] (f16, row-major) x W[N,K]^T (f16, row-major)
// EPI 0: store f32 (no bias);  EPI 1: bias + tanh-GELU -> f16;  EPI 2: bias -> f32
// Block = 4 waves (128 thr). Each wave: one 16-row M tile x NT 16-col N tiles
// (NT accumulators, NT WMMAs per k-step). The NT*16 x 32 weight tile is shared
// by all 4 waves -> staged once per k-step into LDS with the gfx1250 async
// DMA (global_load_async_to_lds_b128 / ASYNCcnt), then ds_load'ed per lane.
// grid = (N/(16*NT), M/64). M mult of 64, N mult of 16*NT, K mult of 32.
// ---------------------------------------------------------------------------
template <int EPI, int NT>
__global__ __launch_bounds__(128)
void gemm_wmma(const _Float16* __restrict__ A, const _Float16* __restrict__ W,
               const float* __restrict__ bias,
               float* __restrict__ Of32, _Float16* __restrict__ Of16,
               int M, int N, int K)
{
    __shared__ __align__(32) _Float16 Bs[NT * 16 * 32];   // NT*16 rows x 32 halves

    const int lane = threadIdx.x & 31;
    const int wave = threadIdx.x >> 5;
    const int n0   = blockIdx.x * (NT * 16);
    const int tm   = blockIdx.y * 4 + wave;      // M tile (16 rows)
    const int hi   = lane >> 4;                  // half-wave select
    const int l16  = lane & 15;

    const _Float16* arow = A + (size_t)(tm * 16 + l16) * K;
    const unsigned ldsbase = (unsigned)(uintptr_t)(&Bs[0]);   // LDS offset = low 32 bits

    v8f acc[NT] = {};

    for (int k0 = 0; k0 < K; k0 += 32) {
        // ---- async-stage W tile: NT*64 chunks of 16B, 128 threads ----
#pragma unroll
        for (int it = 0; it < 2; ++it) {
            int c = threadIdx.x + it * 128;
            if (c < NT * 64) {
                int row = c >> 2, ch = c & 3;                      // 4 x 8-half chunks/row
                const _Float16* g = W + (size_t)(n0 + row) * K + k0 + ch * 8;
                unsigned lds = ldsbase + (unsigned)((row * 32 + ch * 8) * 2);
                asm volatile("global_load_async_to_lds_b128 %0, %1, off"
                             :: "v"(lds), "v"((unsigned long long)(uintptr_t)g)
                             : "memory");
            }
        }
        asm volatile("s_wait_asynccnt 0" ::: "memory");
        __syncthreads();

        // ---- A fragment: halves s -> K = 16*hi + s (matches B) ----
        v8h alo = *(const v8h*)(arow + k0 + hi * 16);
        v8h ahi = *(const v8h*)(arow + k0 + hi * 16 + 8);
        v16h a;
#pragma unroll
        for (int j = 0; j < 8; ++j) { a[j] = alo[j]; a[j + 8] = ahi[j]; }
        if (k0 + 32 < K) __builtin_prefetch(arow + k0 + 32, 0, 1);  // global_prefetch_b8

        // ---- NT back-to-back WMMAs, B fragments from LDS ----
#pragma unroll
        for (int t = 0; t < NT; ++t) {
            // B fragment: column = l16 (+16*t), halves j -> K = 16*hi + j
            v16h b = *(const v16h*)(Bs + (t * 16 + l16) * 32 + hi * 16);
            acc[t] = __builtin_amdgcn_wmma_f32_16x16x32_f16(
                /*neg_a=*/false, a, /*neg_b=*/false, b,
                /*c_mod=*/(short)0, acc[t], /*reuse_a=*/false, /*reuse_b=*/false);
        }
        __syncthreads();   // LDS tile reused next k-step
    }

#pragma unroll
    for (int t = 0; t < NT; ++t) {
        const int ncol = n0 + t * 16 + l16;
#pragma unroll
        for (int r = 0; r < 8; ++r) {
            const int mm = tm * 16 + hi * 8 + r;     // C/D layout: M = 8*hi + r
            float v = acc[t][r];
            if (EPI >= 1) v += bias[ncol];
            if (EPI == 1) {
                float g = 0.5f * v * (1.f + tanhf(0.7978845608f * (v + 0.044715f * v * v * v)));
                Of16[(size_t)mm * N + ncol] = (_Float16)g;
            } else {
                Of32[(size_t)mm * N + ncol] = v;
            }
        }
    }
}

// ---------------------------------------------------------------------------
// fp32 -> f16 conversions
// ---------------------------------------------------------------------------
__global__ void cvt_f16_kernel(const float* __restrict__ src, _Float16* __restrict__ dst, int n)
{
    for (int i = blockIdx.x * blockDim.x + threadIdx.x; i < n; i += gridDim.x * blockDim.x)
        dst[i] = (_Float16)src[i];
}

__global__ void cvt_pair_kernel(const float* __restrict__ src,
                                float* __restrict__ d32, _Float16* __restrict__ d16, int n)
{
    for (int i = blockIdx.x * blockDim.x + threadIdx.x; i < n; i += gridDim.x * blockDim.x) {
        float v = src[i];
        d32[i] = v;
        d16[i] = (_Float16)v;
    }
}

// ---------------------------------------------------------------------------
// Causal depthwise conv (D_CONV=4) + bias + SiLU. xin part of xz = cols [0,512).
// ---------------------------------------------------------------------------
__global__ void conv_silu_kernel(const float* __restrict__ xz,
                                 const float* __restrict__ cw, const float* __restrict__ cb,
                                 float* __restrict__ xin, _Float16* __restrict__ xin16)
{
    int idx = blockIdx.x * blockDim.x + threadIdx.x;
    if (idx >= NTOK * D_INNER) return;
    int d = idx % D_INNER;
    int t = idx / D_INNER;         // token = b*SEQ + l
    int l = t % SEQ;
    float acc = cb[d];
#pragma unroll
    for (int k = 0; k < D_CONV; ++k) {
        int ls = l - (D_CONV - 1) + k;
        if (ls >= 0)
            acc += xz[(size_t)(t + ls - l) * (2 * D_INNER) + d] * cw[d * D_CONV + k];
    }
    float s = acc / (1.f + __expf(-acc));
    xin[idx]   = s;
    xin16[idx] = (_Float16)s;
}

// ---------------------------------------------------------------------------
// dt = softplus(dt_part @ dw^T + db); proj cols [0,16) are dt ranks
// ---------------------------------------------------------------------------
__global__ void dt_kernel(const float* __restrict__ proj, const float* __restrict__ dw,
                          const float* __restrict__ db, float* __restrict__ dt)
{
    int idx = blockIdx.x * blockDim.x + threadIdx.x;
    if (idx >= NTOK * D_INNER) return;
    int d = idx % D_INNER;
    int t = idx / D_INNER;
    const float* p = proj + (size_t)t * (DT_RANK + 2 * D_STATE);
    float acc = db[d];
#pragma unroll
    for (int r = 0; r < DT_RANK; ++r) acc += p[r] * dw[d * DT_RANK + r];
    dt[idx] = (acc > 20.f) ? acc : log1pf(__expf(acc));
}

// ---------------------------------------------------------------------------
// Selective scan. Lane = state n (16 lanes per (b,d) group, 2 groups / wave32).
// dty holds dt on entry, y on exit (in-wave read-before-write, program-ordered).
// Unroll 4: per-step loads are recurrence-independent -> compiler batches them.
// ---------------------------------------------------------------------------
__global__ __launch_bounds__(256)
void scan_kernel(const float* __restrict__ xin, const float* __restrict__ proj,
                 const float* __restrict__ Alog, const float* __restrict__ Dp,
                 float* __restrict__ dty)
{
    int gid = blockIdx.x * 256 + threadIdx.x;
    int n = gid & 15;
    int g = gid >> 4;          // (b,d) group: 0..2047
    int b = g >> 9;
    int d = g & (D_INNER - 1);

    float A  = -__expf(Alog[d * D_STATE + n]);
    float Dv = Dp[d];
    float h  = 0.f;
    const int PW = DT_RANK + 2 * D_STATE;       // 48
    size_t tbase = (size_t)b * SEQ;

#pragma unroll 4
    for (int l = 0; l < SEQ; ++l) {
        size_t t = tbase + l;
        float dtv = dty[t * D_INNER + d];
        float xv  = xin[t * D_INNER + d];
        float Bv  = proj[t * PW + DT_RANK + n];
        float Cv  = proj[t * PW + DT_RANK + D_STATE + n];
        h = __expf(dtv * A) * h + dtv * Bv * xv;
        float part = h * Cv;
        part += __shfl_xor(part, 1, 16);
        part += __shfl_xor(part, 2, 16);
        part += __shfl_xor(part, 4, 16);
        part += __shfl_xor(part, 8, 16);
        if (n == 0) dty[t * D_INNER + d] = part + xv * Dv;   // y overwrites dt
    }
}

// ---------------------------------------------------------------------------
// y * silu(z) -> f16 for out_proj. z = xz cols [512,1024).
// ---------------------------------------------------------------------------
__global__ void gate_silu_kernel(const float* __restrict__ y, const float* __restrict__ xz,
                                 _Float16* __restrict__ yg16)
{
    int idx = blockIdx.x * blockDim.x + threadIdx.x;
    if (idx >= NTOK * D_INNER) return;
    int d = idx % D_INNER;
    int t = idx / D_INNER;
    float z = xz[(size_t)t * (2 * D_INNER) + D_INNER + d];
    float v = y[idx] * (z / (1.f + __expf(-z)));
    yg16[idx] = (_Float16)v;
}

// ---------------------------------------------------------------------------
// x = LayerNorm(mamba_out + residual); block = one token row, LDS reduction
// ---------------------------------------------------------------------------
__global__ __launch_bounds__(256)
void residual_ln_kernel(const float* __restrict__ mout,
                        float* __restrict__ curx, _Float16* __restrict__ curx16,
                        const float* __restrict__ lnw, const float* __restrict__ lnb)
{
    __shared__ float red[D_MODEL];
    int t = blockIdx.x, d = threadIdx.x;
    size_t i = (size_t)t * D_MODEL + d;
    float v = mout[i] + curx[i];
    red[d] = v; __syncthreads();
    for (int s = 128; s > 0; s >>= 1) { if (d < s) red[d] += red[d + s]; __syncthreads(); }
    float mu = red[0] * (1.f / D_MODEL); __syncthreads();
    float c = v - mu;
    red[d] = c * c; __syncthreads();
    for (int s = 128; s > 0; s >>= 1) { if (d < s) red[d] += red[d + s]; __syncthreads(); }
    float var = red[0] * (1.f / D_MODEL);
    float o = c * rsqrtf(var + 1e-5f) * lnw[d] + lnb[d];
    curx[i] = o;
    curx16[i] = (_Float16)o;
}

// ---------------------------------------------------------------------------
// Top-1 gate over 2 experts
// ---------------------------------------------------------------------------
__global__ void moe_gate_kernel(const float* __restrict__ x, const float* __restrict__ gw,
                                const float* __restrict__ gb,
                                int* __restrict__ idx, float* __restrict__ wt)
{
    int t = blockIdx.x * blockDim.x + threadIdx.x;
    if (t >= NTOK) return;
    const float* xr = x + (size_t)t * D_MODEL;
    float l0 = gb[0], l1 = gb[1];
    for (int i = 0; i < D_MODEL; ++i) { float xv = xr[i]; l0 += xv * gw[i]; l1 += xv * gw[D_MODEL + i]; }
    float m = fmaxf(l0, l1);
    float e0 = __expf(l0 - m), e1 = __expf(l1 - m);
    float inv = 1.f / (e0 + e1);
    float p0 = e0 * inv, p1 = e1 * inv;
    int id = (p1 > p0) ? 1 : 0;                  // argmax ties -> expert 0
    idx[t] = id;
    wt[t]  = id ? p1 : p0;
}

__global__ void moe_combine_kernel(const float* __restrict__ curx,
                                   const float* __restrict__ outs0, const float* __restrict__ outs1,
                                   const int* __restrict__ idx, const float* __restrict__ wt,
                                   float* __restrict__ out)
{
    int i = blockIdx.x * blockDim.x + threadIdx.x;
    if (i >= NTOK * D_MODEL) return;
    int t = i >> 8;
    const float* o = idx[t] ? outs1 : outs0;
    out[i] = curx[i] + 0.01f * wt[t] * o[i];
}

// ---------------------------------------------------------------------------
extern "C" void kernel_launch(void* const* d_in, const int* in_sizes, int n_in,
                              void* d_out, int out_size, void* d_ws, size_t ws_size,
                              hipStream_t stream)
{
    const float* x    = (const float*)d_in[0];
    const float* ipw  = (const float*)d_in[1];
    const float* cw   = (const float*)d_in[2];
    const float* cb   = (const float*)d_in[3];
    const float* xpw  = (const float*)d_in[4];
    const float* dtw  = (const float*)d_in[5];
    const float* dtb  = (const float*)d_in[6];
    const float* alog = (const float*)d_in[7];
    const float* dpar = (const float*)d_in[8];
    const float* opw  = (const float*)d_in[9];
    const float* lnw  = (const float*)d_in[10];
    const float* lnb  = (const float*)d_in[11];
    const float* gw   = (const float*)d_in[12];
    const float* gb   = (const float*)d_in[13];
    const float* ew1  = (const float*)d_in[14];
    const float* eb1  = (const float*)d_in[15];
    const float* ew2  = (const float*)d_in[16];
    const float* eb2  = (const float*)d_in[17];
    float* out = (float*)d_out;

    char* wsb = (char*)d_ws;
    size_t off = 0;
    auto alloc = [&](size_t bytes) -> char* {
        char* p = wsb + off;
        off += (bytes + 255) & ~(size_t)255;
        return p;
    };

    _Float16* w16_ip = (_Float16*)alloc((size_t)N_LAYERS * 2 * D_INNER * D_MODEL * 2);
    _Float16* w16_xp = (_Float16*)alloc((size_t)N_LAYERS * (DT_RANK + 2 * D_STATE) * D_INNER * 2);
    _Float16* w16_op = (_Float16*)alloc((size_t)N_LAYERS * D_MODEL * D_INNER * 2);
    _Float16* w16_e1 = (_Float16*)alloc((size_t)N_EXP * D_FF * D_MODEL * 2);
    _Float16* w16_e2 = (_Float16*)alloc((size_t)N_EXP * D_MODEL * D_FF * 2);
    float*    curx   = (float*)   alloc((size_t)NTOK * D_MODEL * 4);
    _Float16* curx16 = (_Float16*)alloc((size_t)NTOK * D_MODEL * 2);
    float*    xz     = (float*)   alloc((size_t)NTOK * 2 * D_INNER * 4);
    float*    xin    = (float*)   alloc((size_t)NTOK * D_INNER * 4);
    _Float16* xin16  = (_Float16*)alloc((size_t)NTOK * D_INNER * 2);
    float*    proj   = (float*)   alloc((size_t)NTOK * (DT_RANK + 2 * D_STATE) * 4);
    float*    dty    = (float*)   alloc((size_t)NTOK * D_INNER * 4);  // dt, then y
    _Float16* yg16   = (_Float16*)alloc((size_t)NTOK * D_INNER * 2);
    float*    mout   = (float*)   alloc((size_t)NTOK * D_MODEL * 4);
    _Float16* h16    = (_Float16*)alloc((size_t)N_EXP * NTOK * D_FF * 2);
    float*    outs   = (float*)   alloc((size_t)N_EXP * NTOK * D_MODEL * 4);
    int*      gidx   = (int*)     alloc((size_t)NTOK * 4);
    float*    gwt    = (float*)   alloc((size_t)NTOK * 4);
    (void)ws_size; (void)in_sizes; (void)n_in; (void)out_size;

    // ---- weight conversion (deterministic each call) ----
    auto cvt = [&](const float* s, _Float16* d, int n) {
        cvt_f16_kernel<<<(n + 1023) / 1024, 256, 0, stream>>>(s, d, n);
    };
    cvt(ipw, w16_ip, N_LAYERS * 2 * D_INNER * D_MODEL);
    cvt(xpw, w16_xp, N_LAYERS * (DT_RANK + 2 * D_STATE) * D_INNER);
    cvt(opw, w16_op, N_LAYERS * D_MODEL * D_INNER);
    cvt(ew1, w16_e1, N_EXP * D_FF * D_MODEL);
    cvt(ew2, w16_e2, N_EXP * D_MODEL * D_FF);
    cvt_pair_kernel<<<2048, 256, 0, stream>>>(x, curx, curx16, NTOK * D_MODEL);

    const int ELT = NTOK * D_INNER;               // 4.2M element grids
    for (int i = 0; i < N_LAYERS; ++i) {
        // in_proj: (8192,256) x (1024,256)^T
        gemm_wmma<0, 4><<<dim3((2 * D_INNER) / 64, NTOK / 64), 128, 0, stream>>>(
            curx16, w16_ip + (size_t)i * 2 * D_INNER * D_MODEL, nullptr,
            xz, nullptr, NTOK, 2 * D_INNER, D_MODEL);
        conv_silu_kernel<<<(ELT + 255) / 256, 256, 0, stream>>>(
            xz, cw + (size_t)i * D_INNER * D_CONV, cb + (size_t)i * D_INNER, xin, xin16);
        // x_proj: (8192,512) x (48,512)^T   (N=48 -> NT=3, one N block)
        gemm_wmma<0, 3><<<dim3(1, NTOK / 64), 128, 0, stream>>>(
            xin16, w16_xp + (size_t)i * (DT_RANK + 2 * D_STATE) * D_INNER, nullptr,
            proj, nullptr, NTOK, DT_RANK + 2 * D_STATE, D_INNER);
        dt_kernel<<<(ELT + 255) / 256, 256, 0, stream>>>(
            proj, dtw + (size_t)i * D_INNER * DT_RANK, dtb + (size_t)i * D_INNER, dty);
        scan_kernel<<<(BATCH * D_INNER * D_STATE) / 256, 256, 0, stream>>>(
            xin, proj, alog + (size_t)i * D_INNER * D_STATE, dpar + (size_t)i * D_INNER, dty);
        gate_silu_kernel<<<(ELT + 255) / 256, 256, 0, stream>>>(dty, xz, yg16);
        // out_proj: (8192,512) x (256,512)^T
        gemm_wmma<0, 4><<<dim3(D_MODEL / 64, NTOK / 64), 128, 0, stream>>>(
            yg16, w16_op + (size_t)i * D_MODEL * D_INNER, nullptr,
            mout, nullptr, NTOK, D_MODEL, D_INNER);
        residual_ln_kernel<<<NTOK, D_MODEL, 0, stream>>>(
            mout, curx, curx16, lnw + (size_t)i * D_MODEL, lnb + (size_t)i * D_MODEL);
    }

    // ---- MoE ----
    moe_gate_kernel<<<(NTOK + 255) / 256, 256, 0, stream>>>(curx, gw, gb, gidx, gwt);
    for (int e = 0; e < N_EXP; ++e) {
        gemm_wmma<1, 4><<<dim3(D_FF / 64, NTOK / 64), 128, 0, stream>>>(
            curx16, w16_e1 + (size_t)e * D_FF * D_MODEL, eb1 + (size_t)e * D_FF,
            nullptr, h16 + (size_t)e * NTOK * D_FF, NTOK, D_FF, D_MODEL);
        gemm_wmma<2, 4><<<dim3(D_MODEL / 64, NTOK / 64), 128, 0, stream>>>(
            h16 + (size_t)e * NTOK * D_FF, w16_e2 + (size_t)e * D_MODEL * D_FF,
            eb2 + (size_t)e * D_MODEL,
            outs + (size_t)e * NTOK * D_MODEL, nullptr, NTOK, D_MODEL, D_FF);
    }
    moe_combine_kernel<<<(NTOK * D_MODEL + 255) / 256, 256, 0, stream>>>(
        curx, outs, outs + (size_t)NTOK * D_MODEL, gidx, gwt, out);
}
